// EnhancedCGCNN_60498909331490
// MI455X (gfx1250) — matrix operand order repaired and checked by reading, not compiled, not run on hardware
//
#include <hip/hip_runtime.h>

// ---------------- model dims ----------------
#define NATOMS (512 * 128)        // 65536 atoms
#define NM     (NATOMS * 8)       // 524288 (atom, neighbor) rows
#define BN_S   0.9999950000374997f
#define LN_EPS 1e-5f

typedef __attribute__((ext_vector_type(8)))  __bf16 v8bf;
typedef __attribute__((ext_vector_type(16))) __bf16 v16bf;
typedef __attribute__((ext_vector_type(8)))  float  v8f;

// round-to-nearest-even float -> bf16 (one-time converts / epilogues only)
static __device__ __forceinline__ __bf16 f2bf(float f) {
  union { float f; unsigned u; } c; c.f = f;
  unsigned r = c.u + 0x7FFFu + ((c.u >> 16) & 1u);
  unsigned short h = (unsigned short)(r >> 16);
  return __builtin_bit_cast(__bf16, h);
}

static __device__ __forceinline__ float sigmoidf_(float x) { return 1.f / (1.f + expf(-x)); }
static __device__ __forceinline__ float softplusf_(float x) { return (x > 20.f) ? x : log1pf(expf(x)); }

static __device__ __forceinline__ v8f wmma_bf16(v16bf a, v16bf b, v8f c) {
  return __builtin_amdgcn_wmma_f32_16x16x32_bf16(false, a, false, b, (short)0, c, false, false);
}

// A fragment (16-bit 16x32, ISA 7.12.2): two contiguous 8-elem runs per lane.
static __device__ __forceinline__ v16bf load_a16(const __bf16* p) {
  v8bf x0 = *(const v8bf*)(p);
  v8bf x1 = *(const v8bf*)(p + 16);
  return __builtin_shufflevector(x0, x1, 0,1,2,3,4,5,6,7,8,9,10,11,12,13,14,15);
}
// B fragment from transposed weights Bt[n][k]: two 8-elem k runs per lane (matches A).
static __device__ __forceinline__ v16bf load_b16(const __bf16* p) {
  v8bf x0 = *(const v8bf*)(p);
  v8bf x1 = *(const v8bf*)(p + 16);
  return __builtin_shufflevector(x0, x1, 0,1,2,3,4,5,6,7,8,9,10,11,12,13,14,15);
}

// ---------------------------------------------------------------------------
// Wide BF16 WMMA GEMM: one wave computes 32 rows x 64 cols (2 row frags
// reused over 4 col tiles -> 8 wmma per 12 b128 loads per K-step).
// A: bf16, K padded to mult of 32.  B: transposed bf16 [Npad x Kpad], Npad>=64.
// mode 0 plain / 1 row-gather / 2 concat of three 64-wide segments.
// act 0 none / 1 relu / 2 relu(x*BN_S).  cstore 0 f32 / 1 bf16(pad->0) / 2 both.
// grid.x = rows/32, grid.y = Npad/64 column strips.
// ---------------------------------------------------------------------------
__global__ __launch_bounds__(32) void gemm_bf16_w(
    const __bf16* __restrict__ A0, const __bf16* __restrict__ A1,
    const __bf16* __restrict__ A2, int ldap, const int* __restrict__ rowidx,
    const __bf16* __restrict__ Bt, int ldbt, const float* __restrict__ bias,
    float* __restrict__ C, __bf16* __restrict__ Cb, int ldc,
    int K, int Ncols, int mode, int act, int cstore) {
  const int lane = threadIdx.x;
  const int hi = lane >> 4;
  const int ln = lane & 15;
  const int rowBase = blockIdx.x * 32;
  const int colBase = blockIdx.y * 64;
  int ar0 = rowBase + ln, ar1 = rowBase + 16 + ln;
  if (mode == 1) { ar0 = rowidx[ar0]; ar1 = rowidx[ar1]; }
  const size_t astr = (mode == 2) ? 64 : (size_t)ldap;
  const __bf16* r0s0 = A0 + (size_t)ar0 * astr;
  const __bf16* r1s0 = A0 + (size_t)ar1 * astr;
  const __bf16* r0s1 = (mode == 2) ? (A1 + (size_t)ar0 * 64) : r0s0;
  const __bf16* r1s1 = (mode == 2) ? (A1 + (size_t)ar1 * 64) : r1s0;
  const __bf16* r0s2 = (mode == 2) ? (A2 + (size_t)ar0 * 64) : r0s0;
  const __bf16* r1s2 = (mode == 2) ? (A2 + (size_t)ar1 * 64) : r1s0;
  const __bf16* bcol = Bt + (size_t)(colBase + ln) * ldbt + hi * 8;

  v8f acc0[4] = {}, acc1[4] = {};
  for (int kb = 0; kb < K; kb += 32) {
    const __bf16 *ap0, *ap1;
    if (mode == 2) {
      if (kb < 64)       { ap0 = r0s0 + kb;        ap1 = r1s0 + kb; }
      else if (kb < 128) { ap0 = r0s1 + (kb - 64); ap1 = r1s1 + (kb - 64); }
      else               { ap0 = r0s2 + (kb - 128); ap1 = r1s2 + (kb - 128); }
    } else { ap0 = r0s0 + kb; ap1 = r1s0 + kb; }
    v16bf af0 = load_a16(ap0 + hi * 8);
    v16bf af1 = load_a16(ap1 + hi * 8);
#pragma unroll
    for (int j = 0; j < 4; ++j) {
      v16bf bf = load_b16(bcol + (size_t)(j * 16) * ldbt + kb);
      acc0[j] = wmma_bf16(af0, bf, acc0[j]);
      acc1[j] = wmma_bf16(af1, bf, acc1[j]);
    }
  }
#pragma unroll
  for (int j = 0; j < 4; ++j) {
    const int n = colBase + j * 16 + ln;
    const bool on = (n < Ncols);
    const float bv = on ? bias[n] : 0.f;
#pragma unroll
    for (int r = 0; r < 8; ++r) {
      int m0 = rowBase + r + hi * 8;        // C layout: vgpr r -> M = r + 8*hi
      int m1 = m0 + 16;
      float v0 = acc0[j][r] + bv, v1 = acc1[j][r] + bv;
      if (act == 1) { v0 = fmaxf(v0, 0.f); v1 = fmaxf(v1, 0.f); }
      else if (act == 2) { v0 = fmaxf(v0 * BN_S, 0.f); v1 = fmaxf(v1 * BN_S, 0.f); }
      if (cstore != 1) {
        if (on) { C[(size_t)m0 * ldc + n] = v0; C[(size_t)m1 * ldc + n] = v1; }
      }
      if (cstore >= 1) {
        Cb[(size_t)m0 * ldc + n] = on ? f2bf(v0) : f2bf(0.f);
        Cb[(size_t)m1 * ldc + n] = on ? f2bf(v1) : f2bf(0.f);
      }
    }
  }
}

// ---------------------------------------------------------------------------
// Fused CGCNN conv layer: one wave = 16 (atom,m) rows x all 128 cols
// (4 filt + 4 core tiles).  The streamed A operand is fetched once per K-step
// and reused across 8 wmma; weights stay cache-hot.  K = 192 padded:
// [x[atom] | x[idx] | nbr_pad], segment select resolved at compile time.
// ---------------------------------------------------------------------------
__global__ __launch_bounds__(32) void conv_bf16(
    const __bf16* __restrict__ xbf, const float* __restrict__ xf,
    const __bf16* __restrict__ nbrbf, const int* __restrict__ nidx,
    const __bf16* __restrict__ Wt /*[128 x 192]*/, const float* __restrict__ bias,
    __bf16* __restrict__ outb) {
  const int lane = threadIdx.x, hi = lane >> 4, ln = lane & 15;
  const int rowBase = blockIdx.x * 16;
  const int rowA = rowBase + ln;
  const int atomA = rowA >> 3;
  const int gA = nidx[rowA];
  const __bf16* px = xbf + (size_t)atomA * 64;
  const __bf16* pg = xbf + (size_t)gA * 64;
  const __bf16* pn = nbrbf + (size_t)rowA * 64;
  const __bf16* bw = Wt + (size_t)ln * 192 + hi * 8;

  v8f aF[4] = {}, aC[4] = {};
#pragma unroll
  for (int s = 0; s < 6; ++s) {
    const int kb = s * 32;
    const __bf16* ap = (s < 2) ? (px + kb) : (s < 4) ? (pg + (kb - 64)) : (pn + (kb - 128));
    v16bf af = load_a16(ap + hi * 8);
#pragma unroll
    for (int t = 0; t < 4; ++t) {
      v16bf bF = load_b16(bw + (size_t)(t * 16) * 192 + kb);
      v16bf bC = load_b16(bw + (size_t)(t * 16 + 64) * 192 + kb);
      aF[t] = wmma_bf16(af, bF, aF[t]);
      aC[t] = wmma_bf16(af, bC, aC[t]);
    }
  }
  const int aOut = 2 * blockIdx.x + hi;  // lanes0-15 -> atom 2b, lanes16-31 -> 2b+1
#pragma unroll
  for (int t = 0; t < 4; ++t) {
    const int colF = t * 16 + ln;
    const float biasF = bias[colF], biasC = bias[colF + 64];
    float sum = 0.f;
#pragma unroll
    for (int r = 0; r < 8; ++r) {
      float f = (aF[t][r] + biasF) * BN_S;
      float c = (aC[t][r] + biasC) * BN_S;
      sum += sigmoidf_(f) * softplusf_(c);
    }
    sum *= BN_S;
    outb[aOut * 64 + colF] = f2bf(softplusf_(xf[aOut * 64 + colF] + sum));
  }
}

// ---------------------------------------------------------------------------
// Attention (faithful to source broadcast M==H): one thread per (n,i).
// ---------------------------------------------------------------------------
__global__ __launch_bounds__(256) void attn_scores(
    const float* __restrict__ q, const float* __restrict__ k,
    const float* __restrict__ v, const float* __restrict__ nf,
    const int* __restrict__ nidx, __bf16* __restrict__ att) {
  int tid = blockIdx.x * blockDim.x + threadIdx.x;
  if (tid >= NM) return;
  int n = tid >> 3, i = tid & 7;
  int a = nidx[n * 8 + i];
  const float* qp = q + n * 64 + i * 8;
  const float* nfp = nf + (size_t)(n * 8 + i) * 64;
  float s[8];
  float mx = -3.4e38f;
#pragma unroll
  for (int j = 0; j < 8; ++j) {
    float d = 0.f;
#pragma unroll
    for (int dd = 0; dd < 8; ++dd)
      d += qp[dd] * (k[a * 64 + j * 8 + dd] + nfp[j * 8 + dd]);
    s[j] = d * 0.35355339059327373f;   // 1/sqrt(8)
    mx = fmaxf(mx, s[j]);
  }
  float ssum = 0.f;
#pragma unroll
  for (int j = 0; j < 8; ++j) { s[j] = expf(s[j] - mx); ssum += s[j]; }
  float inv = 1.f / ssum;
#pragma unroll
  for (int dd = 0; dd < 8; ++dd) {
    float o = 0.f;
#pragma unroll
    for (int j = 0; j < 8; ++j)
      o += s[j] * (v[a * 64 + j * 8 + dd] + nfp[j * 8 + dd]);
    att[n * 64 + i * 8 + dd] = f2bf(o * inv);
  }
}

// Gate blend + LayerNorm; dual-stores fp32 X + bf16 X.
__global__ __launch_bounds__(32) void gate_ln(
    const float* __restrict__ outp, const float* __restrict__ gw,
    const float* __restrict__ gb, const float* __restrict__ lng,
    const float* __restrict__ lnb, float* __restrict__ x,
    __bf16* __restrict__ xb) {
  int row = blockIdx.x, lane = threadIdx.x;
  float o0 = outp[row * 64 + lane], o1 = outp[row * 64 + lane + 32];
  float a0 = x[row * 64 + lane],    a1 = x[row * 64 + lane + 32];
  float gp = o0 * gw[lane] + o1 * gw[lane + 32] + a0 * gw[64 + lane] + a1 * gw[96 + lane];
  for (int off = 16; off; off >>= 1) gp += __shfl_xor(gp, off, 32);
  float g = sigmoidf_(gp + gb[0]);
  float b0 = g * o0 + (1.f - g) * a0, b1 = g * o1 + (1.f - g) * a1;
  float m = b0 + b1;
  for (int off = 16; off; off >>= 1) m += __shfl_xor(m, off, 32);
  m *= (1.f / 64.f);
  float d0 = b0 - m, d1 = b1 - m;
  float vv = d0 * d0 + d1 * d1;
  for (int off = 16; off; off >>= 1) vv += __shfl_xor(vv, off, 32);
  float inv = rsqrtf(vv * (1.f / 64.f) + LN_EPS);
  float r0 = d0 * inv * lng[lane]      + lnb[lane];
  float r1 = d1 * inv * lng[lane + 32] + lnb[lane + 32];
  x[row * 64 + lane]       = r0;
  x[row * 64 + lane + 32]  = r1;
  xb[row * 64 + lane]      = f2bf(r0);
  xb[row * 64 + lane + 32] = f2bf(r1);
}

// Row dot-product for N_out==1 linears, one wave per row.
__global__ __launch_bounds__(32) void rowdot(
    const float* __restrict__ A, int K, const float* __restrict__ w,
    const float* __restrict__ b, float* __restrict__ out) {
  int row = blockIdx.x, lane = threadIdx.x;
  float p = 0.f;
  for (int kk = lane; kk < K; kk += 32) p += A[(size_t)row * K + kk] * w[kk];
  for (int off = 16; off; off >>= 1) p += __shfl_xor(p, off, 32);
  if (lane == 0) out[row] = p + b[0];
}

// Per-crystal softmax over 128 atoms + weighted feature pooling -> bf16 feats.
__global__ __launch_bounds__(128) void pool_softmax(
    const float* __restrict__ s, const float* __restrict__ x,
    const int* __restrict__ cidx, __bf16* __restrict__ feats) {
  __shared__ float w[128];
  __shared__ int ci[128];
  __shared__ float winv;
  int b = blockIdx.x, lv = blockIdx.y, a = threadIdx.x;
  w[a] = s[lv * NATOMS + b * 128 + a];
  ci[a] = cidx[b * 128 + a];
  __syncthreads();
  if (a == 0) {
    float mx = -3.4e38f;
    for (int i = 0; i < 128; ++i) mx = fmaxf(mx, w[i]);
    float sum = 0.f;
    for (int i = 0; i < 128; ++i) { w[i] = expf(w[i] - mx); sum += w[i]; }
    winv = 1.f / sum;
  }
  __syncthreads();
  if (a < 64) {
    float acc = 0.f;
    for (int i = 0; i < 128; ++i) acc += w[i] * x[ci[i] * 64 + a];
    feats[b * 192 + lv * 64 + a] = f2bf(acc * winv);
  }
}

// -------------------- one-time converters --------------------
struct WtJob { const float* W; __bf16* Bt; int K, N, Kpad, Npad; };
struct WtJobs { WtJob j[20]; };

__global__ __launch_bounds__(256) void wt_transpose(WtJobs jobs, int njobs) {
  int jid = blockIdx.y;
  if (jid >= njobs) return;
  WtJob jb = jobs.j[jid];
  int t = blockIdx.x * 256 + threadIdx.x;
  int total = jb.Kpad * jb.Npad;
  if (t >= total) return;
  int k = t % jb.Kpad, n = t / jb.Kpad;
  float v = (k < jb.K && n < jb.N) ? jb.W[k * jb.N + n] : 0.f;
  jb.Bt[(size_t)n * jb.Kpad + k] = f2bf(v);
}

__global__ __launch_bounds__(256) void cvt_pad(
    const float* __restrict__ src, __bf16* __restrict__ dst,
    int rows, int Ksrc, int Kpad) {
  long t = (long)blockIdx.x * 256 + threadIdx.x;
  long total = (long)rows * Kpad;
  if (t >= total) return;
  int k = (int)(t % Kpad);
  long row = t / Kpad;
  float v = (k < Ksrc) ? src[row * Ksrc + k] : 0.f;
  dst[t] = f2bf(v);
}

// ---------------------------------------------------------------------------
// Host side.  Param flatten order (JAX pytree, sorted dict keys) per round 0.
// ---------------------------------------------------------------------------
extern "C" void kernel_launch(void* const* d_in, const int* in_sizes, int n_in,
                              void* d_out, int out_size, void* d_ws, size_t ws_size,
                              hipStream_t stream) {
  (void)in_sizes; (void)n_in; (void)out_size; (void)ws_size;
  const float* atom_fea = (const float*)d_in[0];
  const float* nbr      = (const float*)d_in[1];
  const int*   nidx     = (const int*)d_in[2];
  const int*   cidx     = (const int*)d_in[3];
  auto P = [&](int i) -> const float* { return (const float*)d_in[4 + i]; };

  char* wsb = (char*)d_ws;
  size_t off = 0;
  auto allocF = [&](size_t n) -> float* {
    float* p = (float*)(wsb + off); off += ((n * 4 + 255) & ~(size_t)255); return p;
  };
  auto allocB = [&](size_t n) -> __bf16* {
    __bf16* p = (__bf16*)(wsb + off); off += ((n * 2 + 255) & ~(size_t)255); return p;
  };
  const int N = NATOMS;
  // fp32 activations (elementwise consumers)
  float* X   = allocF((size_t)N * 64);
  float* Q   = allocF((size_t)N * 64);
  float* Kb  = allocF((size_t)N * 64);
  float* V   = allocF((size_t)N * 64);
  float* BE  = allocF((size_t)N * 64);
  float* NF  = allocF((size_t)NM * 64);
  float* L1  = allocF((size_t)N * 32);
  float* S   = allocF((size_t)3 * N);
  float* HB  = allocF((size_t)512 * 128);
  // bf16 activations (WMMA A operands)
  __bf16* Xb    = allocB((size_t)N * 64);
  __bf16* AFb   = allocB((size_t)N * 96);
  __bf16* NBRb  = allocB((size_t)NM * 64);
  __bf16* NT2b  = allocB((size_t)NM * 64);
  __bf16* NT3b  = allocB((size_t)NM * 64);
  __bf16* C0b   = allocB((size_t)N * 64);
  __bf16* C1b   = allocB((size_t)N * 64);
  __bf16* C2b   = allocB((size_t)N * 64);
  __bf16* ATTb  = allocB((size_t)N * 64);
  __bf16* FEATb = allocB((size_t)512 * 192);
  __bf16* CRYb  = allocB((size_t)512 * 64);
  // transposed bf16 weights (Npad >= 64 everywhere)
  __bf16* WEMB = allocB(64 * 96);
  __bf16 *WT2[3], *WT3[3], *WC0[3], *WC1[3], *WC2[3], *WFUS[3],
         *WQ[3], *WK[3], *WV[3], *WO[3], *WNBR[3];
  for (int b = 0; b < 3; ++b) {
    WT2[b] = allocB(64 * 64);   WT3[b] = allocB(64 * 64);
    WC0[b] = allocB(128 * 192); WC1[b] = allocB(128 * 192); WC2[b] = allocB(128 * 192);
    WFUS[b] = allocB(64 * 192);
    WQ[b] = allocB(64 * 64); WK[b] = allocB(64 * 64);
    WV[b] = allocB(64 * 64); WO[b] = allocB(64 * 64);
    WNBR[b] = allocB(64 * 64);
  }
  __bf16* WL1[3];
  for (int l = 0; l < 3; ++l) WL1[l] = allocB(64 * 64);   // Npad 64 (cols 32.. zero)
  __bf16* WPF  = allocB(64 * 192);
  __bf16* WCFC = allocB(128 * 64);

  // ---- one-time weight transposes (batched) ----
  WtJob jl[40]; int nj = 0;
  auto addJ = [&](const float* W, __bf16* Bt, int K, int Nn, int Kp, int Np) {
    jl[nj++] = WtJob{W, Bt, K, Nn, Kp, Np};
  };
  addJ(P(81), WEMB, 92, 64, 96, 64);
  for (int b = 0; b < 3; ++b) {
    int pb = b * 26;
    addJ(P(pb + 23), WT2[b], 41, 41, 64, 64);
    addJ(P(pb + 25), WT3[b], 41, 41, 64, 64);
    addJ(P(pb + 15), WC0[b], 169, 128, 192, 128);
    addJ(P(pb + 17), WC1[b], 169, 128, 192, 128);
    addJ(P(pb + 19), WC2[b], 169, 128, 192, 128);
    addJ(P(pb + 21), WFUS[b], 192, 64, 192, 64);
    addJ(P(pb + 11), WQ[b], 64, 64, 64, 64);
    addJ(P(pb + 3),  WK[b], 64, 64, 64, 64);
    addJ(P(pb + 13), WV[b], 64, 64, 64, 64);
    addJ(P(pb + 9),  WO[b], 64, 64, 64, 64);
    addJ(P(pb + 7),  WNBR[b], 41, 64, 64, 64);
  }
  for (int l = 0; l < 3; ++l) addJ(P(86 + l * 4 + 1), WL1[l], 64, 32, 64, 64);
  addJ(P(85), WPF, 192, 64, 192, 64);
  addJ(P(79), WCFC, 64, 128, 64, 128);
  for (int c = 0; c < nj; c += 20) {
    WtJobs jobs{}; int cnt = (nj - c < 20) ? (nj - c) : 20;
    int mx = 0;
    for (int i = 0; i < cnt; ++i) { jobs.j[i] = jl[c + i];
      int tot = jl[c + i].Kpad * jl[c + i].Npad; if (tot > mx) mx = tot; }
    wt_transpose<<<dim3((mx + 255) / 256, cnt), 256, 0, stream>>>(jobs, cnt);
  }
  // ---- one-time activation converts ----
  cvt_pad<<<dim3((unsigned)(((long)N * 96 + 255) / 256)), 256, 0, stream>>>(atom_fea, AFb, N, 92, 96);
  cvt_pad<<<dim3((unsigned)(((long)NM * 64 + 255) / 256)), 256, 0, stream>>>(nbr, NBRb, NM, 41, 64);

  dim3 blk(32);
  auto gemm = [&](const __bf16* A0, const __bf16* A1, const __bf16* A2, int ldap,
                  const int* ridx, const __bf16* Bt, int ldbt, const float* bias,
                  float* C, __bf16* Cb, int ldc, int rows, int K, int Nc,
                  int nStrips, int mode, int act, int cstore) {
    dim3 grid(rows / 32, nStrips);
    gemm_bf16_w<<<grid, blk, 0, stream>>>(A0, A1, A2, ldap, ridx, Bt, ldbt, bias,
                                          C, Cb, ldc, K, Nc, mode, act, cstore);
  };

  // embedding: relu(lin * BN_S) -> X (f32) + Xb (bf16)
  gemm(AFb, nullptr, nullptr, 96, nullptr, WEMB, 96, P(80), X, Xb, 64, N, 96, 64, 1, 0, 2, 2);

  for (int b = 0; b < 3; ++b) {
    int pb = b * 26;
    // nbr transforms t2/t3 -> bf16 padded [NM x 64]
    gemm(NBRb, nullptr, nullptr, 64, nullptr, WT2[b], 64, P(pb + 22), nullptr, NT2b, 64, NM, 64, 41, 1, 0, 0, 1);
    gemm(NBRb, nullptr, nullptr, 64, nullptr, WT3[b], 64, P(pb + 24), nullptr, NT3b, 64, NM, 64, 41, 1, 0, 0, 1);
    // three fused conv layers -> bf16
    dim3 cgrid(NM / 16);
    conv_bf16<<<cgrid, blk, 0, stream>>>(Xb, X, NBRb, nidx, WC0[b], P(pb + 14), C0b);
    conv_bf16<<<cgrid, blk, 0, stream>>>(Xb, X, NT2b, nidx, WC1[b], P(pb + 16), C1b);
    conv_bf16<<<cgrid, blk, 0, stream>>>(Xb, X, NT3b, nidx, WC2[b], P(pb + 18), C2b);
    // multiscale fusion: relu(concat @ Wf) -> X + Xb
    gemm(C0b, C1b, C2b, 64, nullptr, WFUS[b], 192, P(pb + 20), X, Xb, 64, N, 192, 64, 1, 2, 1, 2);
    // attention projections
    gemm(Xb, nullptr, nullptr, 64, nullptr, WQ[b], 64, P(pb + 10), Q,  nullptr, 64, N, 64, 64, 1, 0, 0, 0);
    gemm(Xb, nullptr, nullptr, 64, nullptr, WK[b], 64, P(pb + 2),  Kb, nullptr, 64, N, 64, 64, 1, 0, 0, 0);
    gemm(Xb, nullptr, nullptr, 64, nullptr, WV[b], 64, P(pb + 12), V,  nullptr, 64, N, 64, 64, 1, 0, 0, 0);
    gemm(NBRb, nullptr, nullptr, 64, nullptr, WNBR[b], 64, P(pb + 6), NF, nullptr, 64, NM, 64, 64, 1, 0, 0, 0);
    attn_scores<<<dim3(NM / 256), dim3(256), 0, stream>>>(Q, Kb, V, NF, nidx, ATTb);
    gemm(ATTb, nullptr, nullptr, 64, nullptr, WO[b], 64, P(pb + 8), BE, nullptr, 64, N, 64, 64, 1, 0, 0, 0);
    gate_ln<<<dim3(N), blk, 0, stream>>>(BE, P(pb + 1), P(pb + 0), P(pb + 5), P(pb + 4), X, Xb);
  }

  // hierarchical pooling
  for (int lv = 0; lv < 3; ++lv) {
    int lb = 86 + lv * 4;
    gemm(Xb, nullptr, nullptr, 64, cidx, WL1[lv], 64, P(lb + 0), L1, nullptr, 32, N, 64, 32, 1, 1, 1, 0);
    rowdot<<<dim3(N), blk, 0, stream>>>(L1, 32, P(lb + 3), P(lb + 2), S + (size_t)lv * N);
  }
  pool_softmax<<<dim3(512, 3), dim3(128), 0, stream>>>(S, X, cidx, FEATb);
  gemm(FEATb, nullptr, nullptr, 192, nullptr, WPF, 192, P(84), nullptr, CRYb, 64, 512, 192, 64, 1, 0, 0, 1);
  gemm(CRYb, nullptr, nullptr, 64, nullptr, WCFC, 64, P(78), HB, nullptr, 128, 512, 64, 128, 2, 0, 2, 0);
  rowdot<<<dim3(512), blk, 0, stream>>>(HB, 128, P(83), P(82), (float*)d_out);
}
